// ResnetVoxelSpherePointHeightnet_63788854280420
// MI455X (gfx1250) — matrix operand order repaired and checked by reading, not compiled, not run on hardware
//
#include <hip/hip_runtime.h>
#include <math.h>
#include <stdint.h>

// ---------------- types for WMMA ----------------
typedef __attribute__((ext_vector_type(16))) _Float16 v16h;
typedef __attribute__((ext_vector_type(8)))  _Float16 v8h;
typedef __attribute__((ext_vector_type(8)))  float    v8f;

union AFrag { v16h v; v8h h[2]; };

// ---------------- problem constants ----------------
#define NB    4
#define NT    65536
#define NBT   (NB * NT)          // 262144 points
#define HID   128
#define XW    256                // activation row width (2*HID)
#define NCELL 4096               // 64*64 spherical cells
#define NR3   (3 * NB * NCELL)   // 49152 rows for channels 0..2
#define KUP   160                // K=131 padded to multiple of 32
#define PI_F  3.1415927410125732f

// monotonic uint mapping for float atomic max/min
__device__ __forceinline__ unsigned int fmap(float f) {
  unsigned int u = __float_as_uint(f);
  return (u & 0x80000000u) ? ~u : (u | 0x80000000u);
}
__device__ __forceinline__ float funmap(unsigned int u) {
  unsigned int b = (u & 0x80000000u) ? (u & 0x7FFFFFFFu) : ~u;
  return __uint_as_float(b);
}

// ---------------- CDNA5 async global->LDS copy (ASYNCcnt path) ----------------
__device__ __forceinline__ void async_b128(unsigned int lds_byte_addr, const void* gptr) {
  asm volatile("global_load_async_to_lds_b128 %0, %1, off"
               :: "v"(lds_byte_addr), "v"((unsigned long long)(uintptr_t)gptr)
               : "memory");
}
__device__ __forceinline__ void wait_async0() {
  asm volatile("s_wait_asynccnt 0x0" ::: "memory");
}

// ---------------- weight prep: f32 [K][N] -> f16 [N][K] (k-major) ----------------
__global__ void prep_w_kt_kernel(const float* __restrict__ w, _Float16* __restrict__ wt,
                                 int K, int N) {
  int idx = blockIdx.x * 256 + threadIdx.x;
  if (idx >= K * N) return;
  int n = idx % N, k = idx / N;
  wt[(size_t)n * K + k] = (_Float16)w[(size_t)k * N + n];
}

__global__ void prep_wu_kernel(const float* __restrict__ wu, _Float16* __restrict__ wut) {
  int idx = blockIdx.x * 256 + threadIdx.x;           // 128 * KUP
  if (idx >= HID * KUP) return;
  int n = idx / KUP, k = idx % KUP;
  wut[idx] = (k < HID + 3) ? (_Float16)wu[(size_t)k * HID + n] : (_Float16)0.0f;
}

// ---------------- x0 = p @ Wp + bp (K=3, trivially VALU) ----------------
__global__ void input_proj_kernel(const float* __restrict__ p, const float* __restrict__ Wp,
                                  const float* __restrict__ bp, _Float16* __restrict__ x) {
  int idx = blockIdx.x * 256 + threadIdx.x;           // NBT * 256
  int i = idx >> 8, j = idx & 255;
  const float* pp = p + (size_t)i * 3;
  float v = bp[j] + pp[0] * Wp[j] + pp[1] * Wp[XW + j] + pp[2] * Wp[2 * XW + j];
  x[(size_t)i * XW + j] = (_Float16)v;
}

__global__ void reset_pooled_kernel(unsigned int* __restrict__ pooled) {
  int idx = blockIdx.x * 256 + threadIdx.x;
  if (idx < NB * HID) pooled[idx] = 0u;               // <= map of any real value
}

// ---------------- ResnetBlockFC stage (WMMA core, LDS-staged weights) ----------------
// x[BT][256] f16 in; writes out tile back into x[:,0:128] (in-place, row-private).
// LDS (32 KB) phase 1: double-buffered W0/Ws k-chunks (2 x (8KB + 8KB)) fed by
// global_load_async_to_lds_b128; phase 2: same 32 KB reused as per-wave h tiles.
__global__ __launch_bounds__(256) void resblock_kernel(
    _Float16* __restrict__ x,
    const _Float16* __restrict__ w0t, const float* __restrict__ b0,   // [128][256], [128]
    const _Float16* __restrict__ w1t, const float* __restrict__ b1,   // [128][128], [128]
    const _Float16* __restrict__ wst,                                 // [128][256]
    unsigned int* __restrict__ pooled, int do_pool) {
  __shared__ __align__(16) char smem[32768];
  const int tid  = threadIdx.x;
  const int lane = tid & 31;
  const int wave = tid >> 5;
  const int hl   = lane & 15;
  const int hi   = lane >> 4;
  const size_t tileRow = ((size_t)blockIdx.x * 8 + wave) * 16;
  const unsigned int lds_base = (unsigned int)(uintptr_t)smem;  // LDS byte address

  const v8f vz = {};
  v8f acc0[8], acc2[8];
#pragma unroll
  for (int nt = 0; nt < 8; ++nt) { acc0[nt] = vz; acc2[nt] = vz; }

  const _Float16* xrow = x + (tileRow + hl) * XW;

  // cooperative async stage of one 32-wide k-chunk of W0 and Ws into buffer `buf`
  // chunk layout: [matrix(0=W0,1=Ws) 8KB][col 0..127][32 halves]
  auto stage_chunk = [&](int buf, int kc) {
    const int m   = tid >> 7;          // 0 = W0, 1 = Ws
    const int col = tid & 127;
    const _Float16* src = (m ? wst : w0t) + (size_t)col * XW + kc;
    const unsigned int dst = lds_base + buf * 16384 + m * 8192 + col * 64;
    async_b128(dst,      src);
    async_b128(dst + 16, src + 8);
    async_b128(dst + 32, src + 16);
    async_b128(dst + 48, src + 24);
  };

  stage_chunk(0, 0);
  wait_async0();
  __syncthreads();

  // GEMM1: relu(x) @ W0   and   GEMM3: x @ Ws   (shared A loads, K=256)
  for (int kc = 0; kc < XW; kc += 32) {
    const int cur = (kc >> 5) & 1;
    if (kc + 32 < XW) stage_chunk(cur ^ 1, kc + 32);

    AFrag xa;
    xa.h[0] = *(const v8h*)(xrow + kc + hi * 16);
    xa.h[1] = *(const v8h*)(xrow + kc + hi * 16 + 8);
    AFrag xr;
#pragma unroll
    for (int e = 0; e < 16; ++e) {
      _Float16 t = xa.v[e];
      xr.v[e] = (t > (_Float16)0.0f) ? t : (_Float16)0.0f;
    }

    const _Float16* c0 = (const _Float16*)(smem + cur * 16384);
    const _Float16* cs = (const _Float16*)(smem + cur * 16384 + 8192);
#pragma unroll
    for (int nt = 0; nt < 8; ++nt) {
      const int col = nt * 16 + hl;
      const _Float16* bp0 = c0 + col * 32 + hi * 16;
      AFrag wf; wf.h[0] = *(const v8h*)bp0; wf.h[1] = *(const v8h*)(bp0 + 8);
      acc0[nt] = __builtin_amdgcn_wmma_f32_16x16x32_f16(
          false, xr.v, false, wf.v, (short)0, acc0[nt], false, false);
      const _Float16* bps = cs + col * 32 + hi * 16;
      AFrag wg; wg.h[0] = *(const v8h*)bps; wg.h[1] = *(const v8h*)(bps + 8);
      acc2[nt] = __builtin_amdgcn_wmma_f32_16x16x32_f16(
          false, xa.v, false, wg.v, (short)0, acc2[nt], false, false);
    }

    if (kc + 32 < XW) wait_async0();   // own async writes done before the barrier
    __syncthreads();                   // chunk reads retired; next buffer visible
  }

  // h = relu(fc0 + b0) -> LDS (chunk buffers are dead; reuse as 8 x 4KB h tiles)
  _Float16* hb = (_Float16*)smem + wave * 16 * HID;
#pragma unroll
  for (int nt = 0; nt < 8; ++nt) {
    const int col = nt * 16 + hl;
    const float bb = b0[col];
#pragma unroll
    for (int r = 0; r < 8; ++r) {
      float v = acc0[nt][r] + bb;
      v = v > 0.0f ? v : 0.0f;
      hb[(hi * 8 + r) * HID + col] = (_Float16)v;
    }
  }

  // GEMM2: h @ W1 (K=128), W1 per-wave from global (L2-resident, 1/5 of the work)
  v8f acc1[8];
#pragma unroll
  for (int nt = 0; nt < 8; ++nt) acc1[nt] = vz;
  const _Float16* hrow = hb + hl * HID;
  for (int kc = 0; kc < HID; kc += 32) {
    AFrag hf;
    hf.h[0] = *(const v8h*)(hrow + kc + hi * 16);
    hf.h[1] = *(const v8h*)(hrow + kc + hi * 16 + 8);
#pragma unroll
    for (int nt = 0; nt < 8; ++nt) {
      const int col = nt * 16 + hl;
      const _Float16* w1p = w1t + (size_t)col * HID + kc + hi * 16;
      __builtin_prefetch(w1p + 32, 0, 1);
      AFrag wf; wf.h[0] = *(const v8h*)w1p; wf.h[1] = *(const v8h*)(w1p + 8);
      acc1[nt] = __builtin_amdgcn_wmma_f32_16x16x32_f16(
          false, hf.v, false, wf.v, (short)0, acc1[nt], false, false);
    }
  }

  // out = x@Ws + (h@W1 + b1); store f16 into x[:,0:128]; pooled atomic max
#pragma unroll
  for (int nt = 0; nt < 8; ++nt) {
    const int col = nt * 16 + hl;
    const float bb = b1[col];
    float mx = -3.402823466e38f;
#pragma unroll
    for (int r = 0; r < 8; ++r) {
      float v = acc2[nt][r] + acc1[nt][r] + bb;
      x[(tileRow + hi * 8 + r) * XW + col] = (_Float16)v;
      mx = fmaxf(mx, v);
    }
    if (do_pool) {
      mx = fmaxf(mx, __shfl_xor(mx, 16, 32));   // partner lane holds the other 8 rows
      if (hi == 0) {
        const int b = (int)(tileRow / NT);
        atomicMax(&pooled[b * HID + col], fmap(mx));
      }
    }
  }
}

// broadcast pooled max into x[:,128:256]
__global__ void concat_pool_kernel(_Float16* __restrict__ x,
                                   const unsigned int* __restrict__ pooled) {
  int idx = blockIdx.x * 256 + threadIdx.x;           // NBT * 128
  int i = idx >> 7, j = idx & 127;
  int b = i >> 16;
  x[(size_t)i * XW + HID + j] = (_Float16)funmap(pooled[b * HID + j]);
}

// ---------------- scatter buffers init ----------------
__global__ void init_scatter_kernel(unsigned int* __restrict__ fmx,
                                    unsigned int* __restrict__ rmn,
                                    unsigned int* __restrict__ rmx,
                                    float* __restrict__ rsum, float* __restrict__ cnt) {
  int idx = blockIdx.x * 256 + threadIdx.x;
  if (idx < 3 * NB * NCELL * HID) fmx[idx] = 0x80000000u;  // map(0.0f): bakes in the clamp
  if (idx < 3 * NB * NCELL) {
    rmn[idx] = 0x80000000u;
    rmx[idx] = 0x80000000u;
    rsum[idx] = 0.0f;
    cnt[idx] = 0.0f;
  }
}

// ---------------- spherical binning scatter ----------------
__global__ void scatter_kernel(const float* __restrict__ p, const _Float16* __restrict__ net,
                               unsigned int* __restrict__ fmx, unsigned int* __restrict__ rmn,
                               unsigned int* __restrict__ rmx, float* __restrict__ rsum,
                               float* __restrict__ cnt) {
  const int i = blockIdx.x * 256 + threadIdx.x;
  if (i >= NBT) return;
  const int b = i >> 16;
  const float px = p[(size_t)i * 3], py = p[(size_t)i * 3 + 1], pz = p[(size_t)i * 3 + 2];
  const float r = sqrtf(px * px + py * py + pz * pz);
  int l = -1;
  if (r < 0.1f) l = 0;
  else if (r > 0.1f && r < 0.3f) l = 1;
  else if (r > 0.3f) l = 2;
  const float lat = 90.0f - atan2f(pz, sqrtf(px * px + py * py)) * (180.0f / PI_F);
  const float mer = fmodf(360.0f + atan2f(py, px) * (180.0f / PI_F), 360.0f);
  const int yg = (int)floorf(lat / (180.0f / 64.0f));
  const int xg = (int)floorf(mer / (360.0f / 64.0f));
  const int cell = xg + 64 * yg;
  for (int ll = 0; ll < 3; ++ll) {
    if (ll == l) {
      if (cell >= 0 && cell < NCELL) {
        const size_t seg = ((size_t)(ll * NB + b)) * NCELL + cell;
        const unsigned int rk = fmap(r);
        atomicMax(&rmx[seg], rk);
        atomicMin(&rmn[seg], rk);
        atomicAdd(&rsum[seg], r);
        atomicAdd(&cnt[seg], 1.0f);
        unsigned int* fb = fmx + seg * HID;
        const _Float16* nrow = net + (size_t)i * XW;
        for (int j = 0; j < HID; ++j)
          atomicMax(&fb[j], fmap((float)nrow[j]));
      }
    } else {
      // masked-out point: pv==0 -> cell W*H/2; only the count is a non-no-op
      const size_t seg0 = ((size_t)(ll * NB + b)) * NCELL + 2048;
      atomicAdd(&cnt[seg0], 1.0f);
    }
  }
}

// ---------------- assemble padded feature matrix [NR3][KUP] f16 ----------------
__global__ void build_feat_kernel(const unsigned int* __restrict__ fmx,
                                  const unsigned int* __restrict__ rmn,
                                  const unsigned int* __restrict__ rmx,
                                  const float* __restrict__ rsum, const float* __restrict__ cnt,
                                  _Float16* __restrict__ feat) {
  const int row = blockIdx.x * 256 + threadIdx.x;
  if (row >= NR3) return;
  const int b = row / (3 * NCELL);
  const int rem = row % (3 * NCELL);
  const int l = rem / NCELL;
  const int cell = rem % NCELL;
  const size_t seg = ((size_t)(l * NB + b)) * NCELL + cell;
  _Float16* fr = feat + (size_t)row * KUP;
  const unsigned int* fb = fmx + seg * HID;
  for (int j = 0; j < HID; ++j) fr[j] = (_Float16)funmap(fb[j]);
  fr[HID + 0] = (_Float16)funmap(rmn[seg]);
  const float c = cnt[seg];
  fr[HID + 1] = (_Float16)(rsum[seg] / fmaxf(c, 1.0f));
  fr[HID + 2] = (_Float16)funmap(rmx[seg]);
  for (int j = HID + 3; j < KUP; ++j) fr[j] = (_Float16)0.0f;
}

// ---------------- final 1x1 conv: feat[NR3][KUP] @ WuT -> out (WMMA) ----------------
__global__ __launch_bounds__(256) void outconv_kernel(const _Float16* __restrict__ feat,
                                                      const _Float16* __restrict__ wut,
                                                      const float* __restrict__ bu,
                                                      float* __restrict__ out) {
  const int lane = threadIdx.x & 31;
  const int wave = threadIdx.x >> 5;
  const int hl = lane & 15;
  const int hi = lane >> 4;
  const size_t tileRow = ((size_t)blockIdx.x * 8 + wave) * 16;

  const v8f vz = {};
  v8f acc[8];
#pragma unroll
  for (int nt = 0; nt < 8; ++nt) acc[nt] = vz;

  const _Float16* frow = feat + (tileRow + hl) * KUP;
  for (int kc = 0; kc < KUP; kc += 32) {
    AFrag af;
    af.h[0] = *(const v8h*)(frow + kc + hi * 16);
    af.h[1] = *(const v8h*)(frow + kc + hi * 16 + 8);
#pragma unroll
    for (int nt = 0; nt < 8; ++nt) {
      const int col = nt * 16 + hl;
      const _Float16* wp = wut + (size_t)col * KUP + kc + hi * 16;
      AFrag wf; wf.h[0] = *(const v8h*)wp; wf.h[1] = *(const v8h*)(wp + 8);
      acc[nt] = __builtin_amdgcn_wmma_f32_16x16x32_f16(
          false, af.v, false, wf.v, (short)0, acc[nt], false, false);
    }
  }

#pragma unroll
  for (int nt = 0; nt < 8; ++nt) {
    const int col = nt * 16 + hl;
    const float bb = bu[col];
#pragma unroll
    for (int r = 0; r < 8; ++r) {
      const size_t row = tileRow + hi * 8 + r;
      const int b = (int)(row / (3 * NCELL));
      const int rem = (int)(row % (3 * NCELL));
      const int l = rem / NCELL;
      const int cell = rem % NCELL;
      out[(((size_t)b * 4 + l) * NCELL + cell) * HID + col] = acc[nt][r] + bb;
    }
  }
}

// channel 3: zero features -> output = bu broadcast
__global__ void fill_ch3_kernel(float* __restrict__ out, const float* __restrict__ bu) {
  const int idx = blockIdx.x * 256 + threadIdx.x;     // NB * NCELL * HID
  if (idx >= NB * NCELL * HID) return;
  const int b = idx / (NCELL * HID);
  const int rem = idx % (NCELL * HID);
  out[(size_t)(b * 4 + 3) * NCELL * HID + rem] = bu[rem % HID];
}

__global__ void zero_tail_kernel(float* __restrict__ out) {
  if (threadIdx.x < 2) out[(size_t)NB * 4 * NCELL * HID + threadIdx.x] = 0.0f;
}

// ---------------- host-side launch ----------------
extern "C" void kernel_launch(void* const* d_in, const int* in_sizes, int n_in,
                              void* d_out, int out_size, void* d_ws, size_t ws_size,
                              hipStream_t stream) {
  (void)in_sizes; (void)n_in; (void)out_size; (void)ws_size;
  const float* p   = (const float*)d_in[0];
  const float* Wp  = (const float*)d_in[1];
  const float* bp  = (const float*)d_in[2];
  const float* W0s = (const float*)d_in[3];
  const float* b0s = (const float*)d_in[4];
  const float* W1s = (const float*)d_in[5];
  const float* b1s = (const float*)d_in[6];
  const float* Wss = (const float*)d_in[7];
  const float* Wu  = (const float*)d_in[8];
  const float* bu  = (const float*)d_in[9];
  float* out = (float*)d_out;

  // --- workspace partition ---
  char* ws = (char*)d_ws;
  size_t off = 0;
  auto alloc = [&](size_t bytes) -> void* {
    void* r = ws + off;
    off = (off + bytes + 255) & ~(size_t)255;
    return r;
  };
  _Float16*     x      = (_Float16*)    alloc((size_t)NBT * XW * 2);          // 128 MiB activations
  _Float16*     w0t    = (_Float16*)    alloc((size_t)5 * HID * XW * 2);
  _Float16*     w1t    = (_Float16*)    alloc((size_t)5 * HID * HID * 2);
  _Float16*     wst    = (_Float16*)    alloc((size_t)5 * HID * XW * 2);
  _Float16*     wut    = (_Float16*)    alloc((size_t)HID * KUP * 2);
  unsigned int* pooled = (unsigned int*)alloc((size_t)NB * HID * 4);
  unsigned int* fmx    = (unsigned int*)alloc((size_t)3 * NB * NCELL * HID * 4);
  unsigned int* rmn    = (unsigned int*)alloc((size_t)3 * NB * NCELL * 4);
  unsigned int* rmx    = (unsigned int*)alloc((size_t)3 * NB * NCELL * 4);
  float*        rsum   = (float*)       alloc((size_t)3 * NB * NCELL * 4);
  float*        cnt    = (float*)       alloc((size_t)3 * NB * NCELL * 4);
  _Float16*     feat   = (_Float16*)    alloc((size_t)NR3 * KUP * 2);

  // --- weight prep (f16, k-major) ---
  for (int s = 0; s < 5; ++s) {
    prep_w_kt_kernel<<<(XW * HID + 255) / 256, 256, 0, stream>>>(
        W0s + (size_t)s * XW * HID, w0t + (size_t)s * HID * XW, XW, HID);
    prep_w_kt_kernel<<<(HID * HID + 255) / 256, 256, 0, stream>>>(
        W1s + (size_t)s * HID * HID, w1t + (size_t)s * HID * HID, HID, HID);
    prep_w_kt_kernel<<<(XW * HID + 255) / 256, 256, 0, stream>>>(
        Wss + (size_t)s * XW * HID, wst + (size_t)s * HID * XW, XW, HID);
  }
  prep_wu_kernel<<<(HID * KUP + 255) / 256, 256, 0, stream>>>(Wu, wut);

  // --- x0 = p @ Wp + bp ---
  input_proj_kernel<<<(NBT * 256) / 256, 256, 0, stream>>>(p, Wp, bp, x);

  // --- 5 resblock stages (WMMA) with global max-pool + concat between ---
  for (int s = 0; s < 5; ++s) {
    const int do_pool = (s < 4) ? 1 : 0;
    if (do_pool)
      reset_pooled_kernel<<<(NB * HID + 255) / 256, 256, 0, stream>>>(pooled);
    resblock_kernel<<<NBT / 128, 256, 0, stream>>>(
        x, w0t + (size_t)s * HID * XW, b0s + s * HID,
        w1t + (size_t)s * HID * HID, b1s + s * HID,
        wst + (size_t)s * HID * XW, pooled, do_pool);
    if (do_pool)
      concat_pool_kernel<<<(NBT * HID) / 256, 256, 0, stream>>>(x, pooled);
  }

  // --- spherical scatter ---
  init_scatter_kernel<<<(3 * NB * NCELL * HID + 255) / 256, 256, 0, stream>>>(
      fmx, rmn, rmx, rsum, cnt);
  scatter_kernel<<<NBT / 256, 256, 0, stream>>>(p, x, fmx, rmn, rmx, rsum, cnt);

  // --- finalize features + 1x1 conv (WMMA) ---
  build_feat_kernel<<<(NR3 + 255) / 256, 256, 0, stream>>>(fmx, rmn, rmx, rsum, cnt, feat);
  outconv_kernel<<<NR3 / 128, 256, 0, stream>>>(feat, wut, bu, out);
  fill_ch3_kernel<<<(NB * NCELL * HID + 255) / 256, 256, 0, stream>>>(out, bu);
  zero_tail_kernel<<<1, 64, 0, stream>>>(out);
}
